// Model_39676907882087
// MI455X (gfx1250) — matrix operand order repaired and checked by reading, not compiled, not run on hardware
//
#include <hip/hip_runtime.h>
#include <stdint.h>

// ---------------------------------------------------------------------------
// Fused attention for MI455X (gfx1250, wave32, WMMA).
//   B=16, LQ=8192, LK=H=128.
// Roofline: ~13 GFLOP of matmul vs ~150 MB HBM traffic (x1 in, mask in, out).
// At bf16 WMMA (v_wmma_f32_16x16x32_bf16, f32 accumulate) the compute drops
// far below the ~6.5 us HBM floor at 23.3 TB/s, so the fused kernel is purely
// memory bound; Q, S, softmax, P never touch HBM.
//
// LDS rows are padded to 136 u16 (272 B = 68 dwords) so that per-lane
// fragment loads (different rows across lanes 0-15) are bank-conflict free:
// dword index = row*68 + c -> bank (4*row + c) mod 64, distinct per lane.
//
// Workspace layout (d_ws, needs ~1.06 MB):
//   [0]                K_bf16  : 16*128*128 u16
//   [16*16384]         V_bf16  : 16*128*128 u16
//   [32*16384]         Wq_bf16 : 128*128    u16
// ---------------------------------------------------------------------------

typedef unsigned short u16;
typedef __attribute__((ext_vector_type(16))) __bf16 v16bf;
typedef __attribute__((ext_vector_type(8)))  float  v8f;

#define BATCH 16
#define LQ    8192
#define LK    128
#define HDIM  128
#define QTILE 128
#define LDSS  136   /* padded LDS row stride in u16 (128 + 8) */
#define INV_SCALE 0.0883883476483184f /* 1/sqrt(128) */

// Hardware RNE f32 -> bf16 (native cvt instruction, not bit twiddling).
__device__ __forceinline__ u16 f32_to_bf16(float f) {
  union { __bf16 h; u16 s; } c; c.h = (__bf16)f; return c.s;
}

// A fragment (16x32 bf16, MxK). Lanes 0-15: M=lane, K = k0+[0..15];
// lanes 16-31: M=lane-16, K = k0+16+[0..15].
__device__ __forceinline__ v16bf load_a_frag(const u16* lds, int row0, int k0) {
  const int lane = threadIdx.x & 31;
  const int hh   = lane >> 4;
  const u16* p = lds + (row0 + (lane & 15)) * LDSS + k0 + hh * 16;
  union { uint4 u[2]; v16bf v; } x;
  x.u[0] = *(const uint4*)(p);
  x.u[1] = *(const uint4*)(p + 8);
  return x.v;
}

// B fragment (32x16 bf16, KxN) read from an [N][K] row-major LDS matrix
// (i.e. B[k][n] = W[n][k]). Lanes 0-15: N=lane, K=k0+[0..15];
// lanes 16-31: N=lane-16, K=k0+16+[0..15]. 32 contiguous bytes per lane.
__device__ __forceinline__ v16bf load_b_frag(const u16* lds, int n0, int k0) {
  const int lane = threadIdx.x & 31;
  const int hh   = lane >> 4;
  const u16* p = lds + (n0 + (lane & 15)) * LDSS + k0 + hh * 16;
  union { uint4 u[2]; v16bf v; } x;
  x.u[0] = *(const uint4*)(p);
  x.u[1] = *(const uint4*)(p + 8);
  return x.v;
}

__device__ __forceinline__ v8f wmma_bf16(v16bf a, v16bf b, v8f c) {
  // (neg_a, A, neg_b, B, c_mod, C, reuse_a, reuse_b)
  return __builtin_amdgcn_wmma_f32_16x16x32_bf16(false, a, false, b,
                                                 (short)0, c, false, false);
}

// Coalesced f32 -> bf16 (packed) staging into padded LDS.
__device__ __forceinline__ void stage_f32_to_lds(const float* __restrict__ src,
                                                 u16* __restrict__ dst) {
  const int t = threadIdx.x;
#pragma unroll
  for (int i = 0; i < 16; ++i) {          // 16384 floats, float4 coalesced
    const int e   = (i * 256 + t) * 4;
    const int row = e >> 7, col = e & 127;
    float4 f = *(const float4*)(src + e);
    uint2 p;
    p.x = f32_to_bf16(f.x) | ((uint32_t)f32_to_bf16(f.y) << 16);
    p.y = f32_to_bf16(f.z) | ((uint32_t)f32_to_bf16(f.w) << 16);
    *(uint2*)(dst + row * LDSS + col) = p;
  }
}

// Coalesced bf16 copy (workspace, stride 128) into padded LDS (stride 136).
__device__ __forceinline__ void stage_bf16_to_lds(const u16* __restrict__ src,
                                                  u16* __restrict__ dst) {
  const int t = threadIdx.x;
  const uint4* s = (const uint4*)src;
  uint4* d = (uint4*)dst;                 // 136 u16 = 17 uint4 per row
#pragma unroll
  for (int i = 0; i < 8; ++i) {           // 2048 uint4 per matrix
    const int g = i * 256 + t;
    d[(g >> 4) * 17 + (g & 15)] = s[g];
  }
}

// ---------------------------------------------------------------------------
// Kernel 1: K = x2*Wk^T, V = x2*Wv^T per batch, stored bf16 in workspace.
// One WG per batch; waves 0-3 compute K, waves 4-7 compute V.
// ---------------------------------------------------------------------------
__global__ __launch_bounds__(256) void kv_proj_kernel(
    const float* __restrict__ x2, const float* __restrict__ wk,
    const float* __restrict__ wv, u16* __restrict__ kbf,
    u16* __restrict__ vbf) {
  extern __shared__ __align__(16) char smem[];
  u16* x2s = (u16*)smem;            // 128 rows x 136 stride
  u16* wks = x2s + HDIM * LDSS;
  u16* wvs = wks + HDIM * LDSS;

  const int b = blockIdx.x;
  stage_f32_to_lds(x2 + (size_t)b * LK * HDIM, x2s);
  stage_f32_to_lds(wk, wks);
  stage_f32_to_lds(wv, wvs);
  __syncthreads();

  const int t    = threadIdx.x;
  const int wave = t >> 5;
  const int lane = t & 31;
  const int hh   = lane >> 4;
  const u16* bsrc = (wave < 4) ? wks : wvs;
  u16* dst = ((wave < 4) ? kbf : vbf) + (size_t)b * LK * HDIM;
  const int rowbase = (wave & 3) * 32;

  for (int rg = 0; rg < 2; ++rg) {
    const int wrow = rowbase + rg * 16;
    for (int nt = 0; nt < 8; ++nt) {
      v8f acc = {0.f, 0.f, 0.f, 0.f, 0.f, 0.f, 0.f, 0.f};
#pragma unroll
      for (int k0 = 0; k0 < HDIM; k0 += 32)
        acc = wmma_bf16(load_a_frag(x2s, wrow, k0),
                        load_b_frag(bsrc, nt * 16, k0), acc);
      const int col = nt * 16 + (lane & 15);
#pragma unroll
      for (int r = 0; r < 8; ++r)
        dst[(wrow + hh * 8 + r) * HDIM + col] = f32_to_bf16(acc[r]);
    }
  }
}

// Kernel 2: Wq f32 -> bf16 workspace copy.
__global__ __launch_bounds__(256) void wq_cvt_kernel(
    const float* __restrict__ wq, u16* __restrict__ wqbf) {
  const int i = blockIdx.x * 256 + threadIdx.x;
  if (i < HDIM * HDIM) wqbf[i] = f32_to_bf16(wq[i]);
}

// ---------------------------------------------------------------------------
// Kernel 3: fused Q-projection + masked softmax attention for one
// (batch, 128-row q-tile). 256 threads = 8 wave32; each wave owns 16 rows.
// ---------------------------------------------------------------------------
__global__ __launch_bounds__(256) void attn_fused_kernel(
    const float* __restrict__ x1, const unsigned char* __restrict__ x3,
    const u16* __restrict__ kbf, const u16* __restrict__ vbf,
    const u16* __restrict__ wqbf, float* __restrict__ out) {
  extern __shared__ __align__(16) char smem[];
  u16* x1s = (u16*)smem;                       // 34 KB  x1 tile (bf16)
  u16* wqs = x1s + HDIM * LDSS;                // 34 KB  Wq
  u16* ks  = wqs + HDIM * LDSS;                // 34 KB  K[b]
  u16* vs  = ks  + HDIM * LDSS;                // 34 KB  V[b]
  u16* qs  = vs  + HDIM * LDSS;                // 34 KB  Q tile
  u16* ps  = qs  + HDIM * LDSS;                // 34 KB  softmax probs
  unsigned char* ms = (unsigned char*)(ps + HDIM * LDSS);  // 16 KB mask

  const int t     = threadIdx.x;
  const int b     = blockIdx.x >> 6;
  const int qbase = (blockIdx.x & 63) * QTILE;

  const unsigned char* x3p = x3 + ((size_t)b * LQ + qbase) * LK;
  __builtin_prefetch(x3p + t * 64, 0, 0);      // pull mask tile toward GL2

  // --- stage x1 tile (f32 -> bf16), Wq/K[b]/V[b] (bf16), mask ---
  stage_f32_to_lds(x1 + ((size_t)b * LQ + qbase) * HDIM, x1s);
  stage_bf16_to_lds(wqbf, wqs);
  stage_bf16_to_lds(kbf + (size_t)b * LK * HDIM, ks);
  stage_bf16_to_lds(vbf + (size_t)b * LK * HDIM, vs);
  {
    const uint4* s = (const uint4*)x3p;
    uint4* d = (uint4*)ms;
#pragma unroll
    for (int i = 0; i < 4; ++i) { const int e = i * 256 + t; d[e] = s[e]; }
  }
  __syncthreads();

  const int wave = t >> 5;
  const int lane = t & 31;
  const int hh   = lane >> 4;      // 0: rows r, 1: rows r+8 (C/D layout)
  const int wrow = wave * 16;
  const int coln = lane & 15;

  // ---- Stage 1: Q = x1 * Wq^T  (B[k][n] = Wq[n][k]) ----
  for (int nt = 0; nt < 8; ++nt) {
    v8f acc = {0.f, 0.f, 0.f, 0.f, 0.f, 0.f, 0.f, 0.f};
#pragma unroll
    for (int k0 = 0; k0 < HDIM; k0 += 32)
      acc = wmma_bf16(load_a_frag(x1s, wrow, k0),
                      load_b_frag(wqs, nt * 16, k0), acc);
    const int col = nt * 16 + coln;
#pragma unroll
    for (int r = 0; r < 8; ++r)
      qs[(wrow + hh * 8 + r) * LDSS + col] = f32_to_bf16(acc[r]);
  }
  __syncthreads();

  // ---- Stage 2: S = Q * K^T, mask, softmax over LK ----
  v8f sacc[8];
  for (int lt = 0; lt < 8; ++lt) {
    v8f acc = {0.f, 0.f, 0.f, 0.f, 0.f, 0.f, 0.f, 0.f};
#pragma unroll
    for (int k0 = 0; k0 < HDIM; k0 += 32)
      acc = wmma_bf16(load_a_frag(qs, wrow, k0),
                      load_b_frag(ks, lt * 16, k0), acc);
    sacc[lt] = acc;
  }
  const float NEG_INF = -__builtin_inff();
#pragma unroll
  for (int r = 0; r < 8; ++r) {
    const int row = wrow + hh * 8 + r;      // q row within tile
    float val[8];
    float rmax = NEG_INF;
#pragma unroll
    for (int lt = 0; lt < 8; ++lt) {
      float sv = sacc[lt][r] * INV_SCALE;
      if (ms[row * LK + lt * 16 + coln]) sv = NEG_INF;   // masked_fill(-inf)
      val[lt] = sv;
      rmax = fmaxf(rmax, sv);
    }
    // one row lives in one 16-lane half: XOR masks 1/2/4/8 stay in-half
#pragma unroll
    for (int off = 1; off < 16; off <<= 1)
      rmax = fmaxf(rmax, __shfl_xor(rmax, off, 32));
    float sum = 0.f;
#pragma unroll
    for (int lt = 0; lt < 8; ++lt) { val[lt] = __expf(val[lt] - rmax); sum += val[lt]; }
#pragma unroll
    for (int off = 1; off < 16; off <<= 1)
      sum += __shfl_xor(sum, off, 32);
    const float inv = 1.0f / sum;
#pragma unroll
    for (int lt = 0; lt < 8; ++lt)
      ps[row * LDSS + lt * 16 + coln] = f32_to_bf16(val[lt] * inv);
  }
  __syncthreads();

  // ---- Stage 3: O = V * P^T  -> out[b, l, qbase+q] ----
  float* outp = out + (size_t)b * HDIM * LQ + qbase;
  for (int qt = 0; qt < 8; ++qt) {
    v8f acc = {0.f, 0.f, 0.f, 0.f, 0.f, 0.f, 0.f, 0.f};
#pragma unroll
    for (int k0 = 0; k0 < HDIM; k0 += 32)
      acc = wmma_bf16(load_a_frag(vs, wrow, k0),
                      load_b_frag(ps, qt * 16, k0), acc);
    const int q = qt * 16 + coln;
#pragma unroll
    for (int r = 0; r < 8; ++r)
      outp[(size_t)(wrow + hh * 8 + r) * LQ + q] = acc[r];
  }
}

// ---------------------------------------------------------------------------
extern "C" void kernel_launch(void* const* d_in, const int* in_sizes, int n_in,
                              void* d_out, int out_size, void* d_ws,
                              size_t ws_size, hipStream_t stream) {
  const float*         x1 = (const float*)d_in[0];
  const float*         x2 = (const float*)d_in[1];
  const unsigned char* x3 = (const unsigned char*)d_in[2];   // bool mask
  const float*         wq = (const float*)d_in[3];
  const float*         wk = (const float*)d_in[4];
  const float*         wv = (const float*)d_in[5];
  float*               out = (float*)d_out;

  u16* kbf  = (u16*)d_ws;                       // 16*128*128 bf16
  u16* vbf  = kbf + (size_t)BATCH * LK * HDIM;  // 16*128*128 bf16
  u16* wqbf = vbf + (size_t)BATCH * LK * HDIM;  // 128*128    bf16

  // Stage K/V projections (bf16) into workspace.
  kv_proj_kernel<<<BATCH, 256, 3 * HDIM * LDSS * sizeof(u16), stream>>>(
      x2, wk, wv, kbf, vbf);
  wq_cvt_kernel<<<(HDIM * HDIM + 255) / 256, 256, 0, stream>>>(wq, wqbf);

  // Fused attention: one WG per (batch, 128-row q-tile) = 16*64 = 1024 WGs.
  const size_t lds_bytes = 6 * HDIM * LDSS * sizeof(u16) + QTILE * LK; // ~220 KB
  attn_fused_kernel<<<BATCH * (LQ / QTILE), 256, lds_bytes, stream>>>(
      x1, x3, kbf, vbf, wqbf, out);
}
